// MultiHeadAttention_16269336117666
// MI455X (gfx1250) — matrix operand order repaired and checked by reading, not compiled, not run on hardware
//
#include <hip/hip_runtime.h>

// ---------------------------------------------------------------------------
// MHA forward for gfx1250 (MI455X): bf16 WMMA, f32 softmax, TDM + async-LDS.
//  d_out = [ output (B*S*D f32) | attn (B*H*S*S f32) ]
//  d_ws  = [ Qp | Kp | Vp | Ctx ]  each B*S*D bf16 (25.2 MB total)
// ---------------------------------------------------------------------------

typedef __bf16    v16bf __attribute__((ext_vector_type(16)));
typedef __bf16    bf2   __attribute__((ext_vector_type(2)));
typedef float     v8f   __attribute__((ext_vector_type(8)));
typedef unsigned  uv4   __attribute__((ext_vector_type(4)));
typedef float     fv4   __attribute__((ext_vector_type(4)));
typedef unsigned  su4   __attribute__((ext_vector_type(4)));   // SGPR quad (TDM D# grp0)
typedef unsigned  su8   __attribute__((ext_vector_type(8)));   // SGPR octet (TDM D# grp1)

constexpr int B_ = 2, S_ = 2048, D_ = 768, H_ = 12, DK_ = 64;
constexpr int M_ = B_ * S_;     // 4096 fused rows of [B*S, D]
constexpr int LDA = 40;         // padded LDS row stride (bf16 elems)

union FragBF {
  unsigned u[8];
  uv4      v[2];
  v16bf    bf;
};

__device__ inline v8f zero8() {
  v8f z;
#pragma unroll
  for (int i = 0; i < 8; ++i) z[i] = 0.f;
  return z;
}

// single hardware v_cvt_pk_bf16_f32 per pair (vector fptrunc)
__device__ inline unsigned pk2(float a, float b) {
  bf2 t;
  t[0] = (__bf16)a;
  t[1] = (__bf16)b;
  return __builtin_bit_cast(unsigned, t);
}
__device__ inline unsigned short f2bf(float f) {
  __bf16 h = (__bf16)f;
  return __builtin_bit_cast(unsigned short, h);
}

__device__ inline v8f bwmma(const FragBF& a, const FragBF& b, v8f c) {
  return __builtin_amdgcn_wmma_f32_16x16x32_bf16(
      /*neg_a=*/false, a.bf, /*neg_b=*/false, b.bf,
      /*c_mod=*/(short)0, c, /*reuse_a=*/false, /*reuse_b=*/false);
}

// ---------------------------------------------------------------------------
// Kernel 1: P = X @ W^T  (torch Linear), bf16 output.
// 64x64 output tile / workgroup (8 waves; wave w -> rows (w&3)*16, 2 col tiles).
// ---------------------------------------------------------------------------
__global__ void __launch_bounds__(256) proj_kernel(
    const float* __restrict__ Xq, const float* __restrict__ Xk, const float* __restrict__ Xv,
    const float* __restrict__ Wq, const float* __restrict__ Wk, const float* __restrict__ Wv,
    unsigned short* __restrict__ Oq, unsigned short* __restrict__ Ok, unsigned short* __restrict__ Ov)
{
  __shared__ unsigned short As[64][LDA];   // 64 rows x 32 k (bf16)
  __shared__ unsigned short Bs[64][LDA];   // 64 n-rows of W x 32 k (bf16)

  const float* X; const float* W; unsigned short* O;
  if (blockIdx.z == 0)      { X = Xq; W = Wq; O = Oq; }
  else if (blockIdx.z == 1) { X = Xk; W = Wk; O = Ok; }
  else                      { X = Xv; W = Wv; O = Ov; }

  const int m0  = blockIdx.x * 64;
  const int n0  = blockIdx.y * 64;
  const int tid = threadIdx.x;
  const int l   = tid & 31;
  const int nl  = l & 15;
  const int hi  = l >> 4;
  const int w   = tid >> 5;
  const int mi  = w & 3;
  const int njb = (w >> 2) * 2;

  const int srow = tid >> 2;
  const int skk  = (tid & 3) * 8;

  v8f acc0 = zero8(), acc1 = zero8();

  for (int k0 = 0; k0 < D_; k0 += 32) {
    __syncthreads();
    {
      const float* p = X + (size_t)(m0 + srow) * D_ + k0 + skk;
      fv4 x0 = *(const fv4*)p, x1 = *(const fv4*)(p + 4);
      uv4 pa; pa[0] = pk2(x0[0], x0[1]); pa[1] = pk2(x0[2], x0[3]);
      pa[2] = pk2(x1[0], x1[1]); pa[3] = pk2(x1[2], x1[3]);
      *(uv4*)&As[srow][skk] = pa;

      const float* pw = W + (size_t)(n0 + srow) * D_ + k0 + skk;
      fv4 w0 = *(const fv4*)pw, w1 = *(const fv4*)(pw + 4);
      uv4 pb; pb[0] = pk2(w0[0], w0[1]); pb[1] = pk2(w0[2], w0[3]);
      pb[2] = pk2(w1[0], w1[1]); pb[3] = pk2(w1[2], w1[3]);
      *(uv4*)&Bs[srow][skk] = pb;
    }
    __syncthreads();

    FragBF a;
    const unsigned short* ar = &As[mi * 16 + nl][0];
    a.v[0] = *(const uv4*)(ar + 16 * hi);
    a.v[1] = *(const uv4*)(ar + 16 * hi + 8);

    FragBF b0, b1;
    const unsigned short* br0 = &Bs[njb * 16 + nl][16 * hi];
    b0.v[0] = *(const uv4*)(br0);
    b0.v[1] = *(const uv4*)(br0 + 8);
    const unsigned short* br1 = &Bs[(njb + 1) * 16 + nl][16 * hi];
    b1.v[0] = *(const uv4*)(br1);
    b1.v[1] = *(const uv4*)(br1 + 8);

    acc0 = bwmma(a, b0, acc0);
    acc1 = bwmma(a, b1, acc1);
  }

#pragma unroll
  for (int r = 0; r < 8; ++r) {
    const int m = m0 + mi * 16 + r + 8 * hi;
    O[(size_t)m * D_ + n0 + njb * 16 + nl]       = f2bf(acc0[r]);
    O[(size_t)m * D_ + n0 + (njb + 1) * 16 + nl] = f2bf(acc1[r]);
  }
}

// ---------------------------------------------------------------------------
// Kernel 2: fused attention. One wave per (16 q-rows, head, batch).
// Pass 1: online (rowmax, expsum); pass 2: write normalized attn, P @ V.
// V tile is fetched by the Tensor Data Mover with LDS padding (144B rows).
// ---------------------------------------------------------------------------
__global__ void __launch_bounds__(32) attn_kernel(
    const unsigned short* __restrict__ Qp,
    const unsigned short* __restrict__ Kp,
    const unsigned short* __restrict__ Vp,
    const int*   __restrict__ mask,
    const float* __restrict__ pbias,
    float*       __restrict__ attn,
    unsigned short* __restrict__ ctx)
{
  __shared__ unsigned short Ps[16][LDA];  // 16x32 P tile (bf16)
  __shared__ unsigned short Vs[32][72];   // 32x64 V tile (bf16), 16B pad per row

  const int l  = threadIdx.x;
  const int nl = l & 15;
  const int hi = l >> 4;
  const int q0 = blockIdx.x * 16;
  const int h  = blockIdx.y;
  const int b  = blockIdx.z;

  const unsigned short* Qb = Qp + (size_t)b * S_ * D_ + h * DK_;
  const unsigned short* Kb = Kp + (size_t)b * S_ * D_ + h * DK_;
  const unsigned short* Vb = Vp + (size_t)b * S_ * D_ + h * DK_;
  const float* pb = pbias + (size_t)h * S_ * S_;
  const int*   mk = mask  + (size_t)b * S_ * S_;
  float*       at = attn  + ((size_t)b * H_ + h) * S_ * S_;

  // Q as A operand: 16 rows x DK=64 -> two 16x32 fragments
  FragBF aq0, aq1;
  {
    const unsigned short* qr = Qb + (size_t)(q0 + nl) * D_;
    aq0.v[0] = *(const uv4*)(qr + 16 * hi);
    aq0.v[1] = *(const uv4*)(qr + 16 * hi + 8);
    aq1.v[0] = *(const uv4*)(qr + 32 + 16 * hi);
    aq1.v[1] = *(const uv4*)(qr + 32 + 16 * hi + 8);
  }

  float lm[8], ls[8];
#pragma unroll
  for (int r = 0; r < 8; ++r) { lm[r] = -3.0e38f; ls[r] = 0.f; }

  // ---------------- pass 1 ----------------
  for (int n0 = 0; n0 < S_; n0 += 16) {
    if (n0 + 16 < S_)
      __builtin_prefetch(Kb + (size_t)(n0 + 16 + nl) * D_, 0, 1);  // global_prefetch_b8
    const unsigned short* kr = Kb + (size_t)(n0 + nl) * D_;
    FragBF bk0, bk1;
    bk0.v[0] = *(const uv4*)(kr + 16 * hi);
    bk0.v[1] = *(const uv4*)(kr + 16 * hi + 8);
    bk1.v[0] = *(const uv4*)(kr + 32 + 16 * hi);
    bk1.v[1] = *(const uv4*)(kr + 32 + 16 * hi + 8);
    v8f c = zero8();
    c = bwmma(aq0, bk0, c);
    c = bwmma(aq1, bk1, c);
    const int kcol = n0 + nl;
#pragma unroll
    for (int r = 0; r < 8; ++r) {
      const int q = q0 + r + 8 * hi;
      float s = c[r] * 0.125f + pb[(size_t)q * S_ + kcol];
      if (mk[(size_t)q * S_ + kcol] == 0) s = -1e9f;
      float nm = fmaxf(lm[r], s);
      ls[r] = ls[r] * __expf(lm[r] - nm) + __expf(s - nm);
      lm[r] = nm;
    }
  }

  // combine (max,sum) across the 16 lanes sharing each row group
#pragma unroll
  for (int r = 0; r < 8; ++r) {
#pragma unroll
    for (int off = 1; off < 16; off <<= 1) {
      float om = __shfl_xor(lm[r], off, 32);
      float os = __shfl_xor(ls[r], off, 32);
      float nm = fmaxf(lm[r], om);
      ls[r] = ls[r] * __expf(lm[r] - nm) + os * __expf(om - nm);
      lm[r] = nm;
    }
    ls[r] = 1.0f / ls[r];   // 1/rowsum
  }

  v8f cc0 = zero8(), cc1 = zero8(), cc2 = zero8(), cc3 = zero8();

  const unsigned ldsVbase = (unsigned)(size_t)&Vs[0][0];

  // ---------------- pass 2 ----------------
  for (int n0 = 0; n0 < S_; n0 += 32) {
    // TDM fetch of the 32x64 V tile, overlapped with QK^T recompute.
    // 2D descriptor, data_size=2B, tile 64x32, tensor stride D=768 elems,
    // LDS pad 4 DWORDs per 32 DWORDs (-> 144B LDS rows = Vs[32][72]).
    {
      unsigned long long ga =
          (unsigned long long)(size_t)(const void*)(Vb + (size_t)n0 * D_);
      su4 g0;
      g0[0] = 1u;                                   // count=1 valid descriptor
      g0[1] = ldsVbase;                             // lds_addr
      g0[2] = (unsigned)ga;                         // global_addr[31:0]
      g0[3] = (unsigned)((ga >> 32) & 0x01FFFFFFull) | (2u << 30);  // addr[56:32] | type=2
      su8 g1;
      g1[0] = (1u << 16)      // data_size = 2 bytes
            | (1u << 20)      // pad_enable
            | (4u << 22)      // pad_interval: every 32 DWORDs (128B)
            | (3u << 25);     // pad_amount: 4 DWORDs (16B)
      g1[1] = (64u << 16);    // tensor_dim0 = 64 (lo16 in [31:16])
      g1[2] = (32u << 16);    // tensor_dim0 hi=0 | tensor_dim1 = 32 (lo16)
      g1[3] = (64u << 16);    // tensor_dim1 hi=0 | tile_dim0 = 64
      g1[4] = 32u;            // tile_dim1 = 32, tile_dim2 = 0
      g1[5] = 768u;           // tensor_dim0_stride = D (lo32)
      g1[6] = 0u;
      g1[7] = 0u;
      asm volatile("tensor_load_to_lds %0, %1" :: "s"(g0), "s"(g1) : "memory");
    }

#pragma unroll
    for (int t = 0; t < 2; ++t) {
      const int nb = n0 + t * 16;
      const unsigned short* kr = Kb + (size_t)(nb + nl) * D_;
      FragBF bk0, bk1;
      bk0.v[0] = *(const uv4*)(kr + 16 * hi);
      bk0.v[1] = *(const uv4*)(kr + 16 * hi + 8);
      bk1.v[0] = *(const uv4*)(kr + 32 + 16 * hi);
      bk1.v[1] = *(const uv4*)(kr + 32 + 16 * hi + 8);
      v8f c = zero8();
      c = bwmma(aq0, bk0, c);
      c = bwmma(aq1, bk1, c);
      const int kcol = nb + nl;
#pragma unroll
      for (int r = 0; r < 8; ++r) {
        const int q = q0 + r + 8 * hi;
        float s = c[r] * 0.125f + pb[(size_t)q * S_ + kcol];
        if (mk[(size_t)q * S_ + kcol] == 0) s = -1e9f;
        float p = __expf(s - lm[r]) * ls[r];
        __builtin_nontemporal_store(p, &at[(size_t)q * S_ + kcol]);  // 403MB stream
        Ps[r + 8 * hi][t * 16 + nl] = f2bf(p);
      }
    }

    __builtin_amdgcn_s_wait_tensorcnt(0);   // V tile landed in LDS
    __syncthreads();                        // single-wave WG: S_NOP, orders DS

    FragBF ap;
    const unsigned short* pr = &Ps[nl][0];
    ap.v[0] = *(const uv4*)(pr + 16 * hi);
    ap.v[1] = *(const uv4*)(pr + 16 * hi + 8);

#pragma unroll
    for (int dkt = 0; dkt < 4; ++dkt) {
      FragBF bv;
      const int dk = dkt * 16 + nl;
#pragma unroll
      for (int i = 0; i < 8; ++i) {
        unsigned lo = Vs[16 * hi + 2 * i][dk];
        unsigned hh = Vs[16 * hi + 2 * i + 1][dk];
        bv.u[i] = lo | (hh << 16);
      }
      v8f& cc = (dkt == 0) ? cc0 : (dkt == 1) ? cc1 : (dkt == 2) ? cc2 : cc3;
      cc = bwmma(ap, bv, cc);
    }
    __syncthreads();
  }

#pragma unroll
  for (int r = 0; r < 8; ++r) {
    const int q = q0 + r + 8 * hi;
    unsigned short* cr = ctx + ((size_t)b * S_ + q) * D_ + h * DK_;
    cr[0 * 16 + nl] = f2bf(cc0[r]);
    cr[1 * 16 + nl] = f2bf(cc1[r]);
    cr[2 * 16 + nl] = f2bf(cc2[r]);
    cr[3 * 16 + nl] = f2bf(cc3[r]);
  }
}

// ---------------------------------------------------------------------------
// Kernel 3: output = ctx @ Wo^T + bo (f32 out). ctx tile staged with
// global_load_async_to_lds_b128 (pure bf16 copy, ASYNCcnt-tracked).
// ---------------------------------------------------------------------------
__global__ void __launch_bounds__(256) out_kernel(
    const unsigned short* __restrict__ Xc,
    const float* __restrict__ Wo,
    const float* __restrict__ bo,
    float* __restrict__ Out)
{
  __shared__ unsigned short As[64][LDA];
  __shared__ unsigned short Bs[64][LDA];

  const int m0  = blockIdx.x * 64;
  const int n0  = blockIdx.y * 64;
  const int tid = threadIdx.x;
  const int l   = tid & 31;
  const int nl  = l & 15;
  const int hi  = l >> 4;
  const int w   = tid >> 5;
  const int mi  = w & 3;
  const int njb = (w >> 2) * 2;
  const int srow = tid >> 2;
  const int skk  = (tid & 3) * 8;

  v8f acc0 = zero8(), acc1 = zero8();

  for (int k0 = 0; k0 < D_; k0 += 32) {
    __syncthreads();
    {
      // A tile (already bf16): async global->LDS copy, 16B per thread
      unsigned ldsA = (unsigned)(size_t)&As[srow][skk];
      const unsigned short* gp = Xc + (size_t)(m0 + srow) * D_ + k0 + skk;
      asm volatile("global_load_async_to_lds_b128 %0, %1, off"
                   :: "v"(ldsA), "v"(gp) : "memory");

      const float* pw = Wo + (size_t)(n0 + srow) * D_ + k0 + skk;
      fv4 w0 = *(const fv4*)pw, w1 = *(const fv4*)(pw + 4);
      uv4 pbk; pbk[0] = pk2(w0[0], w0[1]); pbk[1] = pk2(w0[2], w0[3]);
      pbk[2] = pk2(w1[0], w1[1]); pbk[3] = pk2(w1[2], w1[3]);
      *(uv4*)&Bs[srow][skk] = pbk;

      asm volatile("s_wait_asynccnt 0" ::: "memory");
    }
    __syncthreads();

    FragBF a;
    const unsigned short* ar = &As[mi * 16 + nl][0];
    a.v[0] = *(const uv4*)(ar + 16 * hi);
    a.v[1] = *(const uv4*)(ar + 16 * hi + 8);

    FragBF b0, b1;
    const unsigned short* br0 = &Bs[njb * 16 + nl][16 * hi];
    b0.v[0] = *(const uv4*)(br0);
    b0.v[1] = *(const uv4*)(br0 + 8);
    const unsigned short* br1 = &Bs[(njb + 1) * 16 + nl][16 * hi];
    b1.v[0] = *(const uv4*)(br1);
    b1.v[1] = *(const uv4*)(br1 + 8);

    acc0 = bwmma(a, b0, acc0);
    acc1 = bwmma(a, b1, acc1);
  }

#pragma unroll
  for (int r = 0; r < 8; ++r) {
    const int m  = m0 + mi * 16 + r + 8 * hi;
    const int n1 = n0 + njb * 16 + nl;
    const int n2 = n0 + (njb + 1) * 16 + nl;
    Out[(size_t)m * D_ + n1] = acc0[r] + bo[n1];
    Out[(size_t)m * D_ + n2] = acc1[r] + bo[n2];
  }
}

// ---------------------------------------------------------------------------
extern "C" void kernel_launch(void* const* d_in, const int* in_sizes, int n_in,
                              void* d_out, int out_size, void* d_ws, size_t ws_size,
                              hipStream_t stream) {
  (void)in_sizes; (void)n_in; (void)out_size; (void)ws_size;
  const float* q  = (const float*)d_in[0];
  const float* k  = (const float*)d_in[1];
  const float* v  = (const float*)d_in[2];
  const int*   mk = (const int*)d_in[3];
  const float* pb = (const float*)d_in[4];
  const float* Wq = (const float*)d_in[5];
  const float* Wk = (const float*)d_in[6];
  const float* Wv = (const float*)d_in[7];
  const float* Wo = (const float*)d_in[8];
  const float* bo = (const float*)d_in[9];

  const size_t bsd = (size_t)B_ * S_ * D_;
  unsigned short* Qp  = (unsigned short*)d_ws;
  unsigned short* Kp  = Qp + bsd;
  unsigned short* Vp  = Kp + bsd;
  unsigned short* Ctx = Vp + bsd;

  float* out  = (float*)d_out;
  float* attn = out + bsd;

  proj_kernel<<<dim3(M_ / 64, D_ / 64, 3), 256, 0, stream>>>(
      q, k, v, Wq, Wk, Wv, Qp, Kp, Vp);
  attn_kernel<<<dim3(S_ / 16, H_, B_), 32, 0, stream>>>(
      Qp, Kp, Vp, mk, pb, attn, Ctx);
  out_kernel<<<dim3(M_ / 64, D_ / 64, 1), 256, 0, stream>>>(
      Ctx, Wo, bo, out);
}
